// SatGNN_5076651344239
// MI455X (gfx1250) — matrix operand order repaired and checked by reading, not compiled, not run on hardware
//
#include <hip/hip_runtime.h>
#include <hip/hip_bf16.h>
#include <math.h>

// ---------------- problem constants (match reference) ----------------
constexpr int kNV  = 50000;
constexpr int kNC  = 20000;
constexpr int kH   = 256;
constexpr int kL   = 4;
constexpr int kEVV = 500000;
constexpr int kEVC = 300000;
constexpr int kG   = 64;

typedef __attribute__((ext_vector_type(16))) __bf16 v16bf;
typedef __attribute__((ext_vector_type(8)))  float  v8f;

__device__ __forceinline__ unsigned short f2bf(float f) {
  unsigned int u = __float_as_uint(f);
  u += 0x7FFFu + ((u >> 16) & 1u);          // round-to-nearest-even
  return (unsigned short)(u >> 16);
}

__device__ __forceinline__ void atomicMaxFloat(float* addr, float val) {
  int iv = __float_as_int(val);
  if (iv >= 0) atomicMax((int*)addr, iv);
  else         atomicMin((unsigned int*)addr, (unsigned int)iv);
}

// ---------------- weight pre-pack: W_src -> WMMA B-fragment layout ----------------
// Per matrix m (256x256): tiles t = nt*8+kt ; per tile 32 lanes x 16 bf16.
// Element e of lane L maps to B[k][n] with k = kt*32 + 8*(L>>4) + (e&7) + 16*(e>>3), n = nt*16 + (L&15).
__global__ void pack_weights(const float* __restrict__ W, unsigned short* __restrict__ Bp) {
  int id = blockIdx.x * 256 + threadIdx.x;          // < 16 * 65536
  int m    = id >> 16;
  int rem  = id & 65535;
  int tile = rem >> 9;                               // nt*8 + kt
  int lane = (rem >> 4) & 31;
  int e    = rem & 15;
  int nt = tile >> 3, kt = tile & 7;
  int half = lane >> 4, nloc = lane & 15;
  int k = kt * 32 + half * 8 + (e & 7) + ((e >> 3) << 4);
  int n = nt * 16 + nloc;
  Bp[id] = f2bf(W[(size_t)m * 65536 + (size_t)k * 256 + n]);
}

// ---------------- fused attention vectors: (W @ a) for all 32 (layer,head) ----------------
__global__ void fuse_att(const float* __restrict__ Wsrc, const float* __restrict__ Wdst,
                         const float* __restrict__ as_,  const float* __restrict__ ad_,
                         float* __restrict__ ws_as, float* __restrict__ wd_ad) {
  int id  = blockIdx.x * 256 + threadIdx.x;          // < 2*32*256
  int sel = id >> 13;
  int r   = id & 8191;
  int m = r >> 8, i = r & 255;
  const float* W = (sel ? Wdst : Wsrc) + (size_t)m * 65536 + (size_t)i * 256;
  const float* a = (sel ? ad_  : as_)  + (size_t)m * 256;
  float s = 0.f;
  #pragma unroll 8
  for (int j = 0; j < 256; ++j) s += W[j] * a[j];
  (sel ? wd_ad : ws_as)[m * 256 + i] = s;
}

// ---------------- f32 -> bf16 activation convert ----------------
__global__ void cvt_bf16(const float* __restrict__ X, unsigned short* __restrict__ Y, int n) {
  int i = blockIdx.x * 256 + threadIdx.x;
  if (i < n) Y[i] = f2bf(X[i]);
}

// ---------------- WMMA GEMM: C(Nx256,f32) = A(Nx256,bf16 row-major) x Bpacked(256x256) ----
// Block = 256 threads = 8 waves; block owns 16 rows x all 256 cols; wave owns 2 n-tiles.
__global__ void __launch_bounds__(256) gemm_xl(const unsigned short* __restrict__ Abf,
                                               const unsigned short* __restrict__ Bp,
                                               float* __restrict__ C, int N) {
  const int wave = threadIdx.x >> 5;
  const int lane = threadIdx.x & 31;
  const int half = lane >> 4;
  const int mloc = lane & 15;                        // A row-in-tile / D column-in-tile
  const int rowBase = blockIdx.x * 16;
  const int nt0 = wave * 2, nt1 = wave * 2 + 1;

  union Frag { uint4 q[2]; v16bf v; };
  v8f acc0 = {}; v8f acc1 = {};

  #pragma unroll
  for (int kt = 0; kt < 8; ++kt) {
    Frag a, b0, b1;
    // A fragment: lane holds row (rowBase+mloc), K = kt*32 + half*8 .. +7 and +16..+23
    const uint4* Aq = (const uint4*)(Abf + (size_t)(rowBase + mloc) * 256 + kt * 32 + half * 8);
    a.q[0] = Aq[0];
    a.q[1] = Aq[2];                                  // +16 bf16 elements
    // B fragments: pre-packed, 16 contiguous bf16 per lane per tile
    const uint4* B0 = (const uint4*)(Bp + ((size_t)((nt0 * 8 + kt) * 32 + lane) * 16));
    b0.q[0] = B0[0]; b0.q[1] = B0[1];
    const uint4* B1 = (const uint4*)(Bp + ((size_t)((nt1 * 8 + kt) * 32 + lane) * 16));
    b1.q[0] = B1[0]; b1.q[1] = B1[1];

    acc0 = __builtin_amdgcn_wmma_f32_16x16x32_bf16(false, a.v, false, b0.v, (short)0, acc0, false, false);
    acc1 = __builtin_amdgcn_wmma_f32_16x16x32_bf16(false, a.v, false, b1.v, (short)0, acc1, false, false);
  }
  // D layout: lane column = mloc, rows = half*8 + r
  #pragma unroll
  for (int r = 0; r < 8; ++r) {
    int row = rowBase + half * 8 + r;
    C[(size_t)row * 256 + nt0 * 16 + mloc] = acc0[r];
    C[(size_t)row * 256 + nt1 * 16 + mloc] = acc1[r];
  }
}

// ---------------- per-row dot with a 256-vector (attention logits) ----------------
__global__ void rowdot(const float* __restrict__ X, const float* __restrict__ v,
                       float* __restrict__ out, int N) {
  int w    = (blockIdx.x * 256 + threadIdx.x) >> 5;
  int lane = threadIdx.x & 31;
  if (w >= N) return;
  const float* xr = X + (size_t)w * 256;
  float s = 0.f;
  #pragma unroll
  for (int j = 0; j < 8; ++j) s += xr[lane + 32 * j] * v[lane + 32 * j];
  #pragma unroll
  for (int off = 16; off; off >>= 1) s += __shfl_down(s, off, 32);
  if (lane == 0) out[w] = s;
}

// ---------------- edge softmax machinery ----------------
__global__ void init_maxden(float* __restrict__ m, float* __restrict__ den, int N) {
  int i = blockIdx.x * 256 + threadIdx.x;
  if (i < N) { m[i] = -INFINITY; den[i] = 0.f; }
}

__global__ void edge_score_max(const int* __restrict__ src, const int* __restrict__ dst,
                               const float* __restrict__ asrc, const float* __restrict__ adst,
                               float* __restrict__ ebuf, float* __restrict__ m, int E) {
  int i = blockIdx.x * 256 + threadIdx.x;
  if (i >= E) return;
  float v = asrc[src[i]] + adst[dst[i]];
  float e = v > 0.f ? v : 0.2f * v;                  // leaky_relu(0.2)
  ebuf[i] = e;
  atomicMaxFloat(&m[dst[i]], e);
}

__global__ void finite_fix(float* __restrict__ m, int N) {
  int i = blockIdx.x * 256 + threadIdx.x;
  if (i < N) { float v = m[i]; if (!isfinite(v)) m[i] = 0.f; }
}

__global__ void edge_exp_sum(float* __restrict__ ebuf, const int* __restrict__ dst,
                             const float* __restrict__ m, float* __restrict__ den, int E) {
  int i = blockIdx.x * 256 + threadIdx.x;
  if (i >= E) return;
  int d = dst[i];
  float ex = __expf(ebuf[i] - m[d]);
  ebuf[i] = ex;
  atomicAdd(&den[d], ex);
}

__global__ void init_bias(float* __restrict__ out, const float* __restrict__ b0,
                          const float* __restrict__ b1, const float* __restrict__ b2, int N) {
  size_t id = (size_t)blockIdx.x * 256 + threadIdx.x;
  if (id >= (size_t)N * 256) return;
  int c = (int)(id & 255);
  float v = b0[c];
  if (b1) v += b1[c];
  if (b2) v += b2[c];
  out[id] = v;
}

// one wave per edge: out[dst] += (ex/den[dst]) * XL[src]  (atomics resolve in L2)
__global__ void edge_aggregate(const float* __restrict__ ebuf, const int* __restrict__ src,
                               const int* __restrict__ dst, const float* __restrict__ den,
                               const float* __restrict__ XL, float* __restrict__ out, int E) {
  int w    = (blockIdx.x * 256 + threadIdx.x) >> 5;
  int lane = threadIdx.x & 31;
  if (w >= E) return;
  int s = src[w], d = dst[w];
  float coeff = ebuf[w] / (den[d] + 1e-16f);
  const float* xs = XL + (size_t)s * 256;
  float* od = out + (size_t)d * 256;
  #pragma unroll
  for (int j = 0; j < 8; ++j) { int c = lane + 32 * j; atomicAdd(&od[c], coeff * xs[c]); }
}

// ---------------- pooling + head ----------------
__global__ void zero_f32(float* __restrict__ p, int n) {
  int i = blockIdx.x * 256 + threadIdx.x;
  if (i < n) p[i] = 0.f;
}

__global__ void pool_accum(const float* __restrict__ X, const int* __restrict__ batch,
                           float* __restrict__ pooled, float* __restrict__ counts,
                           int N, int chanOff) {
  int w    = (blockIdx.x * 256 + threadIdx.x) >> 5;
  int lane = threadIdx.x & 31;
  if (w >= N) return;
  int g = batch[w];
  const float* xr = X + (size_t)w * 256;
  float* pg = pooled + (size_t)g * 512 + chanOff;
  #pragma unroll
  for (int j = 0; j < 8; ++j) { int c = lane + 32 * j; atomicAdd(&pg[c], xr[c]); }
  if (lane == 0) atomicAdd(&counts[g], 1.f);
}

__global__ void __launch_bounds__(256) final_softmax(const float* __restrict__ pooled,
                                                     const float* __restrict__ counts,
                                                     const float* __restrict__ Wlin,
                                                     const float* __restrict__ blin,
                                                     float* __restrict__ out) {
  __shared__ float s0[256], s1[256];
  int g = blockIdx.x, t = threadIdx.x;
  float p0 = 0.f, p1 = 0.f;
  for (int k = t; k < 512; k += 256) {
    float cnt = counts[(k < 256 ? 0 : kG) + g];
    cnt = cnt > 1.f ? cnt : 1.f;
    float v = pooled[(size_t)g * 512 + k] / cnt;
    p0 += v * Wlin[k * 2 + 0];
    p1 += v * Wlin[k * 2 + 1];
  }
  s0[t] = p0; s1[t] = p1; __syncthreads();
  for (int off = 128; off; off >>= 1) {
    if (t < off) { s0[t] += s0[t + off]; s1[t] += s1[t + off]; }
    __syncthreads();
  }
  if (t == 0) {
    float z0 = s0[0] + blin[0], z1 = s1[0] + blin[1];
    float mx = z0 > z1 ? z0 : z1;
    float e0 = __expf(z0 - mx), e1 = __expf(z1 - mx);
    float inv = 1.f / (e0 + e1);
    out[g * 2 + 0] = e0 * inv;
    out[g * 2 + 1] = e1 * inv;
  }
}

// =======================================================================
extern "C" void kernel_launch(void* const* d_in, const int* in_sizes, int n_in,
                              void* d_out, int out_size, void* d_ws, size_t ws_size,
                              hipStream_t stream) {
  (void)in_sizes; (void)n_in; (void)out_size; (void)ws_size;
  const float* x_var   = (const float*)d_in[0];
  const float* x_con   = (const float*)d_in[1];
  const float* W_src   = (const float*)d_in[2];
  const float* W_dst   = (const float*)d_in[3];
  const float* att_src = (const float*)d_in[4];
  const float* att_dst = (const float*)d_in[5];
  const float* bias    = (const float*)d_in[6];
  const float* W_lin   = (const float*)d_in[7];
  const float* b_lin   = (const float*)d_in[8];
  const int*   e_vv    = (const int*)d_in[9];
  const int*   e_vvr   = (const int*)d_in[10];
  const int*   e_vc    = (const int*)d_in[11];
  const int*   e_cv    = (const int*)d_in[12];
  const int*   batch_v = (const int*)d_in[13];
  const int*   batch_c = (const int*)d_in[14];
  float* out = (float*)d_out;

  // ---- workspace carve (bump allocator, 256B aligned) ----
  char* p = (char*)d_ws;
  auto alloc = [&](size_t bytes) -> void* {
    void* r = (void*)p; p += (bytes + 255) & ~(size_t)255; return r;
  };
  unsigned short* Wpack = (unsigned short*)alloc((size_t)16 * 65536 * 2);
  float* ws_as = (float*)alloc(32 * 256 * 4);
  float* wd_ad = (float*)alloc(32 * 256 * 4);
  float* vbufA = (float*)alloc((size_t)kNV * 256 * 4);
  float* vbufB = (float*)alloc((size_t)kNV * 256 * 4);
  float* cbufA = (float*)alloc((size_t)kNC * 256 * 4);
  float* cbufB = (float*)alloc((size_t)kNC * 256 * 4);
  unsigned short* xv_bf = (unsigned short*)alloc((size_t)kNV * 256 * 2);
  unsigned short* xc_bf = (unsigned short*)alloc((size_t)kNC * 256 * 2);
  float* XL    = (float*)alloc((size_t)kNV * 256 * 4);
  float* asrc  = (float*)alloc((size_t)kNV * 4);
  float* adst  = (float*)alloc((size_t)kNV * 4);
  float* mbuf  = (float*)alloc((size_t)kNV * 4);
  float* den   = (float*)alloc((size_t)kNV * 4);
  float* ebuf  = (float*)alloc((size_t)kEVV * 4);
  float* pooled  = (float*)alloc((size_t)kG * 512 * 4);
  float* counts  = (float*)alloc((size_t)2 * kG * 4);

  // ---- one-time (per call) weight prep ----
  pack_weights<<<(16 * 65536) / 256, 256, 0, stream>>>(W_src, Wpack);
  fuse_att<<<(2 * 32 * 256) / 256, 256, 0, stream>>>(W_src, W_dst, att_src, att_dst, ws_as, wd_ad);

  auto run_head = [&](int m, const unsigned short* Xbf, const float* Xsrc, int Nsrc,
                      const float* Xdst, int Ndst, const int* sidx, const int* didx,
                      int E, float* outbuf) {
    gemm_xl<<<Nsrc / 16, 256, 0, stream>>>(Xbf, Wpack + (size_t)m * 65536, XL, Nsrc);
    rowdot<<<(Nsrc * 32 + 255) / 256, 256, 0, stream>>>(Xsrc, ws_as + m * 256, asrc, Nsrc);
    rowdot<<<(Ndst * 32 + 255) / 256, 256, 0, stream>>>(Xdst, wd_ad + m * 256, adst, Ndst);
    init_maxden<<<(Ndst + 255) / 256, 256, 0, stream>>>(mbuf, den, Ndst);
    edge_score_max<<<(E + 255) / 256, 256, 0, stream>>>(sidx, didx, asrc, adst, ebuf, mbuf, E);
    finite_fix<<<(Ndst + 255) / 256, 256, 0, stream>>>(mbuf, Ndst);
    edge_exp_sum<<<(E + 255) / 256, 256, 0, stream>>>(ebuf, didx, mbuf, den, E);
    edge_aggregate<<<(E + 7) / 8, 256, 0, stream>>>(ebuf, sidx, didx, den, XL, outbuf, E);
  };

  const float* cv = x_var;
  const float* cc = x_con;
  for (int l = 0; l < kL; ++l) {
    float* nv = (l & 1) ? vbufB : vbufA;
    float* nc = (l & 1) ? cbufB : cbufA;
    cvt_bf16<<<(kNV * 256 + 255) / 256, 256, 0, stream>>>(cv, xv_bf, kNV * 256);
    cvt_bf16<<<(kNC * 256 + 255) / 256, 256, 0, stream>>>(cc, xc_bf, kNC * 256);
    // init outputs with (summed) biases
    init_bias<<<(kNV * 256 + 255) / 256, 256, 0, stream>>>(
        nv, bias + ((size_t)l * 4 + 0) * kH, bias + ((size_t)l * 4 + 1) * kH,
        bias + ((size_t)l * 4 + 3) * kH, kNV);
    init_bias<<<(kNC * 256 + 255) / 256, 256, 0, stream>>>(
        nc, bias + ((size_t)l * 4 + 2) * kH, nullptr, nullptr, kNC);
    // head 0: v->v  (e_vv)
    run_head(l * 4 + 0, xv_bf, cv, kNV, cv, kNV, e_vv, e_vv + kEVV, kEVV, nv);
    // head 1: v->v reversed (e_vv_rev)
    run_head(l * 4 + 1, xv_bf, cv, kNV, cv, kNV, e_vvr, e_vvr + kEVV, kEVV, nv);
    // head 3: c->v  (e_cv: src=con, dst=var)
    run_head(l * 4 + 3, xc_bf, cc, kNC, cv, kNV, e_cv, e_cv + kEVC, kEVC, nv);
    // head 2: v->c  (e_vc: src=var, dst=con)
    run_head(l * 4 + 2, xv_bf, cv, kNV, cc, kNC, e_vc, e_vc + kEVC, kEVC, nc);
    cv = nv; cc = nc;
  }

  // ---- mean pool + linear + softmax ----
  zero_f32<<<(kG * 512 + 255) / 256, 256, 0, stream>>>(pooled, kG * 512);
  zero_f32<<<1, 256, 0, stream>>>(counts, 2 * kG);
  pool_accum<<<(kNV * 32 + 255) / 256, 256, 0, stream>>>(cv, batch_v, pooled, counts, kNV, 0);
  pool_accum<<<(kNC * 32 + 255) / 256, 256, 0, stream>>>(cc, batch_c, pooled, counts + kG, kNC, 256);
  final_softmax<<<kG, 256, 0, stream>>>(pooled, counts, W_lin, b_lin, out);
}
